// LinearBase_21354577395892
// MI455X (gfx1250) — 8-bit matrix operand order repaired and checked by reading, compiled, not run on hardware
//
#include <hip/hip_runtime.h>


typedef int   v16i __attribute__((ext_vector_type(16)));
typedef float v8f  __attribute__((ext_vector_type(8)));

#define M_DIM 4096
#define K_DIM 4096
#define N_DIM 16384

#define BM 128
#define BN 128
#define BK 128
#define NT (K_DIM / BK)
#define LDS_STRIDE 144

#if defined(__has_builtin)
#  if __has_builtin(__builtin_amdgcn_global_load_async_to_lds_b128)
#    define HAVE_ASYNC_LDS 1
#  endif
#endif
#ifndef HAVE_ASYNC_LDS
#  define HAVE_ASYNC_LDS 0
#endif

#if HAVE_ASYNC_LDS
typedef int vb128 __attribute__((__vector_size__(16)));
typedef __attribute__((address_space(1))) vb128* g_b128p;
typedef __attribute__((address_space(3))) vb128* l_b128p;
#  if __has_builtin(__builtin_amdgcn_s_wait_asynccnt)
#    define WAIT_ASYNC(n) __builtin_amdgcn_s_wait_asynccnt(n)
#  else
#    define WAIT_ASYNC(n) asm volatile("s_wait_asynccnt %0" ::"i"(n) : "memory")
#  endif
#else
#  define WAIT_ASYNC(n) ((void)0)
#endif

union FragAB {
    v16i v;
    unsigned long long u64[8];
};

__device__ __forceinline__ unsigned f32_to_e4m3(float x) {
    unsigned u = __float_as_uint(x);
    unsigned s = (u >> 31) << 7;
    unsigned a = u & 0x7FFFFFFFu;
    if (a >= 0x7F800000u) return s | 0x7Eu;
    float ax = __uint_as_float(a);
    if (ax >= 448.0f) return s | 0x7Eu;
    if (ax < 0.015625f) {
        float t = ax * 512.0f + 12582912.0f;
        unsigned m = __float_as_uint(t) & 0xFu;
        return s | m;
    }
    unsigned r = a + 0x0007FFFFu + ((a >> 20) & 1u);
    int e = (int)(r >> 23) - 127;
    unsigned mant = (r >> 20) & 7u;
    return s | ((unsigned)(e + 7) << 3) | mant;
}

__device__ __forceinline__ unsigned pack4_e4m3(float a, float b, float c, float d) {
    return f32_to_e4m3(a) | (f32_to_e4m3(b) << 8) |
           (f32_to_e4m3(c) << 16) | (f32_to_e4m3(d) << 24);
}

__global__ void __launch_bounds__(256)
quant_x_kernel(const float* __restrict__ x, unsigned char* __restrict__ xq,
               float* __restrict__ xs) {
    const int row = blockIdx.x;
    const int tid = threadIdx.x;
    const float4* xr = (const float4*)(x + (size_t)row * K_DIM);

    float4 v[4];
    float mx = 0.0f;
#pragma unroll
    for (int i = 0; i < 4; ++i) {
        v[i] = xr[tid + i * 256];
        mx = fmaxf(mx, fmaxf(fmaxf(fabsf(v[i].x), fabsf(v[i].y)),
                             fmaxf(fabsf(v[i].z), fabsf(v[i].w))));
    }
    __shared__ float red[256];
    red[tid] = mx;
    __syncthreads();
    for (int sft = 128; sft > 0; sft >>= 1) {
        if (tid < sft) red[tid] = fmaxf(red[tid], red[tid + sft]);
        __syncthreads();
    }
    const float scale = fmaxf(red[0] * (1.0f / 448.0f), 1e-12f);
    if (tid == 0) xs[row] = scale;
    const float inv = 1.0f / scale;

    unsigned* out = (unsigned*)(xq + (size_t)row * K_DIM);
#pragma unroll
    for (int i = 0; i < 4; ++i)
        out[tid + i * 256] =
            pack4_e4m3(v[i].x * inv, v[i].y * inv, v[i].z * inv, v[i].w * inv);
}

__global__ void __launch_bounds__(256)
quant_w_kernel(const float* __restrict__ w, unsigned char* __restrict__ wq) {
    const size_t i = (size_t)blockIdx.x * 256 + threadIdx.x;
    const float4 v = ((const float4*)w)[i];
    ((unsigned*)wq)[i] = pack4_e4m3(v.x, v.y, v.z, v.w);
}

__device__ __forceinline__ void
stage_tile(unsigned char* __restrict__ dstA, unsigned char* __restrict__ dstB,
           const unsigned char* __restrict__ xq, const unsigned char* __restrict__ wq,
           int rowM0, int colN0, int kt, int tid) {
#pragma unroll
    for (int s = 0; s < 4; ++s) {
        const int c   = tid + s * 256;
        const int r   = c >> 3;
        const int col = (c & 7) << 4;
        const unsigned char* ga = &xq[(size_t)(rowM0 + r) * K_DIM + kt + col];
        const unsigned char* gb = &wq[(size_t)(colN0 + r) * K_DIM + kt + col];
        unsigned char* la = &dstA[r * LDS_STRIDE + col];
        unsigned char* lb = &dstB[r * LDS_STRIDE + col];
#if HAVE_ASYNC_LDS
        __builtin_amdgcn_global_load_async_to_lds_b128(
            (g_b128p)(void*)ga, (l_b128p)(void*)la, 0, 0);
        __builtin_amdgcn_global_load_async_to_lds_b128(
            (g_b128p)(void*)gb, (l_b128p)(void*)lb, 0, 0);
#else
        *(uint4*)la = *(const uint4*)ga;
        *(uint4*)lb = *(const uint4*)gb;
#endif
    }
}

__global__ void __launch_bounds__(256)
gemm_fp8_kernel(const unsigned char* __restrict__ xq,
                const unsigned char* __restrict__ wq,
                const float* __restrict__ xs,
                const float* __restrict__ wscale,
                const float* __restrict__ bias,
                float* __restrict__ out) {
    __shared__ __align__(16) unsigned char As[2][BM * LDS_STRIDE];
    __shared__ __align__(16) unsigned char Bs[2][BN * LDS_STRIDE];

    const int tid  = threadIdx.x;
    const int lane = tid & 31;
    const int wave = tid >> 5;
    const int wm   = wave >> 2;
    const int wn   = wave & 3;
    const int half = lane >> 4;
    const int l16  = lane & 15;

    const int rowM0 = blockIdx.y * BM;
    const int colN0 = blockIdx.x * BN;

    v8f acc[4][2];
#pragma unroll
    for (int i = 0; i < 4; ++i)
#pragma unroll
        for (int j = 0; j < 2; ++j)
            acc[i][j] = (v8f){0.f, 0.f, 0.f, 0.f, 0.f, 0.f, 0.f, 0.f};

    stage_tile(As[0], Bs[0], xq, wq, rowM0, colN0, 0, tid);

    for (int t = 0; t < NT; ++t) {
        const int cur = t & 1;
        if (t + 1 < NT) {
            stage_tile(As[cur ^ 1], Bs[cur ^ 1], xq, wq, rowM0, colN0,
                       (t + 1) * BK, tid);
            WAIT_ASYNC(8);
        } else {
            WAIT_ASYNC(0);
        }
        __syncthreads();

        FragAB a[4], b[2];
#pragma unroll
        for (int j = 0; j < 2; ++j) {
            const unsigned char* base =
                &Bs[cur][(wn * 32 + j * 16 + l16) * LDS_STRIDE + half * 16];
#pragma unroll
            for (int q = 0; q < 8; ++q)
                b[j].u64[q] = *(const unsigned long long*)
                                  (base + (q >> 1) * 32 + (q & 1) * 8);
        }
#pragma unroll
        for (int i = 0; i < 4; ++i) {
            const unsigned char* base =
                &As[cur][(wm * 64 + i * 16 + l16) * LDS_STRIDE + half * 16];
#pragma unroll
            for (int p = 0; p < 8; ++p)
                a[i].u64[p] = *(const unsigned long long*)(base + (p >> 1) * 32 + (p & 1) * 8);
        }

#pragma unroll
        for (int i = 0; i < 4; ++i)
#pragma unroll
            for (int j = 0; j < 2; ++j)
                acc[i][j] = __builtin_amdgcn_wmma_f32_16x16x128_fp8_fp8(
                    a[i].v, b[j].v, (short)0, acc[i][j], false, false);

        __syncthreads();
    }

#pragma unroll
    for (int i = 0; i < 4; ++i) {
        const int mBase = rowM0 + wm * 64 + i * 16 + half * 8;
        float xsv[8];
#pragma unroll
        for (int v = 0; v < 8; ++v) xsv[v] = xs[mBase + v];
#pragma unroll
        for (int j = 0; j < 2; ++j) {
            const int n    = colN0 + wn * 32 + j * 16 + l16;
            const float w  = wscale[n];
            const float bn = bias[n];
#pragma unroll
            for (int v = 0; v < 8; ++v)
                out[(size_t)(mBase + v) * N_DIM + n] =
                    acc[i][j][v] * xsv[v] * w + bn;
        }
    }
}

extern "C" void kernel_launch(void* const* d_in, const int* in_sizes, int n_in,
                              void* d_out, int out_size, void* d_ws, size_t ws_size,
                              hipStream_t stream) {
    const float* x      = (const float*)d_in[0];
    const float* w      = (const float*)d_in[1];
    const float* wscale = (const float*)d_in[2];
    const float* bias   = (const float*)d_in[3];
    float* out          = (float*)d_out;

    unsigned char* xq = (unsigned char*)d_ws;
    unsigned char* wq = xq + (size_t)M_DIM * K_DIM;
    float*         xs = (float*)(wq + (size_t)N_DIM * K_DIM);

    quant_x_kernel<<<M_DIM, 256, 0, stream>>>(x, xq, xs);
    quant_w_kernel<<<(int)((size_t)N_DIM * K_DIM / 1024), 256, 0, stream>>>(w, wq);

    dim3 grid(N_DIM / BN, M_DIM / BM);
    gemm_fp8_kernel<<<grid, 256, 0, stream>>>(xq, wq, xs, wscale, bias, out);
}
